// HierarchicalMBCL_81724637708495
// MI455X (gfx1250) — compile-verified, matrix operand order checked by reading, not run on hardware
//
#include <hip/hip_runtime.h>
#include <hip/hip_fp16.h>

typedef __attribute__((ext_vector_type(16))) _Float16 v16h;
typedef __attribute__((ext_vector_type(8)))  float    v8f;

#define D_DIM     128
#define TAU_INV   10.0f
#define HARD_K    32
#define CAP       48
#define NEG_INF   (-3.0e38f)
#define MAX_ROWS  20000

// ---------------- workspace layout (bytes) ----------------
#define OFF_BITMAP 0u           // 100352 B  (100000 user flags, padded)
#define OFF_META_N 100352u      // int[8]   per-pair common counts
#define OFF_META_S 100416u      // float[8] per-pair loss sums
#define OFF_COMMON 100608u      // int[20000]
#define OFF_AW     262144u      // f16 [20000][128] weak, WMMA-swizzled
#define OFF_AS     5644288u     // f16 [20000][128] strong, WMMA-swizzled

// ---------------- phase kernels ----------------
__global__ void k_zero(unsigned int* bm_words, int nWords, int* pairN, float* pairSum, int p) {
    int i = blockIdx.x * blockDim.x + threadIdx.x;
    if (i < nWords) bm_words[i] = 0u;
    if (i == 0) { pairN[p] = 0; pairSum[p] = 0.0f; }
}

__global__ void k_mark(const long long* uS, int n, unsigned char* bitmap) {
    int i = blockIdx.x * blockDim.x + threadIdx.x;
    if (i < n) bitmap[(int)uS[i]] = 1;
}

__global__ void k_compact(const long long* uW, int n, const unsigned char* bitmap,
                          int* commonIdx, int* pairN, int p) {
    int i = blockIdx.x * blockDim.x + threadIdx.x;
    if (i < n) {
        int u = (int)uW[i];
        if (bitmap[u]) {
            int pos = atomicAdd(&pairN[p], 1);
            commonIdx[pos] = u;
        }
    }
}

// Gather, L2-normalize, convert to f16 and store in the per-lane WMMA fragment
// order: each 32-wide K-chunk stored as [K0..7, K16..23 | K8..15, K24..31] so a
// lane's v16h fragment is one 32-byte contiguous read. Rows padded (zero) to a
// multiple of 32 for the M=32 strips of the main kernel.
__global__ void __launch_bounds__(128) k_gather(const float* __restrict__ embW,
                                                const float* __restrict__ embS,
                                                const int* __restrict__ commonIdx,
                                                const int* __restrict__ pairN, int p,
                                                _Float16* __restrict__ Aw,
                                                _Float16* __restrict__ As) {
    int r = blockIdx.x;
    int N = pairN[p];
    int Npad = ((N + 31) >> 5) << 5;
    if (r >= Npad) return;

    int t   = threadIdx.x;          // 0..127 == K index
    int kc  = t >> 5;               // 32-wide K chunk
    int rk  = t & 31;
    int sub = rk >> 3;
    int posc = rk + (sub == 1 ? 8 : (sub == 2 ? -8 : 0));
    size_t dst = (size_t)r * D_DIM + kc * 32 + posc;

    if (r >= N) { Aw[dst] = (_Float16)0.0f; As[dst] = (_Float16)0.0f; return; }

    int u = commonIdx[r];
    float w = embW[(size_t)u * D_DIM + t];
    float s = embS[(size_t)u * D_DIM + t];

    __shared__ float red[2][4];
    float sw = w * w, ss = s * s;
    for (int off = 16; off > 0; off >>= 1) {
        sw += __shfl_xor(sw, off, 32);
        ss += __shfl_xor(ss, off, 32);
    }
    if ((t & 31) == 0) { red[0][kc] = sw; red[1][kc] = ss; }
    __syncthreads();
    sw = red[0][0] + red[0][1] + red[0][2] + red[0][3];
    ss = red[1][0] + red[1][1] + red[1][2] + red[1][3];
    float nw = fmaxf(sqrtf(sw), 1e-12f);
    float ns = fmaxf(sqrtf(ss), 1e-12f);
    Aw[dst] = (_Float16)(w / nw);
    As[dst] = (_Float16)(s / ns);
}

// Wave-cooperative exact top-32 compaction of one row's candidate buffer.
__device__ __forceinline__ void topk_compact(float* row, int* pcnt, float* pthr, int lane) {
    __shared__ float tmpk[HARD_K];
    int c = *pcnt; if (c > CAP) c = CAP;
    for (int i = 0; i < HARD_K; ++i) {
        float v1 = (lane < c)      ? row[lane]      : NEG_INF;
        float v2 = (lane + 32 < c) ? row[lane + 32] : NEG_INF;
        float mv = v1; int mi = lane;
        if (v2 > mv) { mv = v2; mi = lane + 32; }
        for (int off = 16; off > 0; off >>= 1) {
            float ov = __shfl_xor(mv, off, 32);
            int   oi = __shfl_xor(mi, off, 32);
            if (ov > mv) { mv = ov; mi = oi; }
        }
        if (lane == 0) { tmpk[i] = mv; row[mi] = NEG_INF; }
        __syncthreads();
    }
    if (lane < HARD_K) row[lane] = tmpk[lane];
    if (lane == 0) { *pcnt = HARD_K; *pthr = tmpk[HARD_K - 1]; }
    __syncthreads();
}

// One wave per 32-row strip: fused f16 WMMA GEMM (8 WMMAs per 16-col tile,
// register double-buffered B stream) + streaming exact top-32 + logsumexp.
__global__ void __launch_bounds__(32) k_simloss(const _Float16* __restrict__ Aw,
                                                const _Float16* __restrict__ As,
                                                const int* __restrict__ pairN,
                                                float* __restrict__ pairSum, int p) {
    __shared__ float buf[32][CAP];
    __shared__ float posv[32];
    __shared__ float thr[32];
    __shared__ int   cnt[32];

    int N = pairN[p];
    int rowBase = blockIdx.x << 5;
    if (N < 4 || rowBase >= N) return;

    int lane = threadIdx.x;
    cnt[lane] = 0; thr[lane] = NEG_INF; posv[lane] = NEG_INF;
    __syncthreads();

    int g  = lane >> 4;     // K-group of this lane
    int lm = lane & 15;     // row/col within 16-tile

    // Two A-fragment sets: rows [rowBase, +16) and [rowBase+16, +32).
    const _Float16* arow0 = As + (size_t)(rowBase + lm) * D_DIM + g * 16;
    const _Float16* arow1 = arow0 + 16 * D_DIM;
    v16h a0 = *(const v16h*)(arow0 +  0);
    v16h a1 = *(const v16h*)(arow0 + 32);
    v16h a2 = *(const v16h*)(arow0 + 64);
    v16h a3 = *(const v16h*)(arow0 + 96);
    v16h a4 = *(const v16h*)(arow1 +  0);
    v16h a5 = *(const v16h*)(arow1 + 32);
    v16h a6 = *(const v16h*)(arow1 + 64);
    v16h a7 = *(const v16h*)(arow1 + 96);

    // Loop-invariant per-row state kept in registers (16 rows per lane).
    bool  rowOk[16];
    float thrv[16];
    #pragma unroll
    for (int v = 0; v < 8; ++v) {
        rowOk[v]     = (rowBase + v + (g << 3)) < N;
        rowOk[v + 8] = (rowBase + 16 + v + (g << 3)) < N;
        thrv[v] = NEG_INF; thrv[v + 8] = NEG_INF;
    }
    bool needThr = false;

    int nTiles = (N + 15) >> 4;
    size_t laneOff = (size_t)lm * D_DIM + g * 16;

    // Preload B fragments for tile 0.
    const _Float16* brow = Aw + laneOff;
    v16h b0 = *(const v16h*)(brow +  0);
    v16h b1 = *(const v16h*)(brow + 32);
    v16h b2 = *(const v16h*)(brow + 64);
    v16h b3 = *(const v16h*)(brow + 96);

    for (int ct = 0; ct < nTiles; ++ct) {
        int colBase = ct << 4;

        // Issue next tile's loads BEFORE using the current fragments.
        int ctn = (ct + 1 < nTiles) ? ct + 1 : ct;   // clamp: stay in-bounds
        const _Float16* brown = Aw + (size_t)(ctn << 4) * D_DIM + laneOff;
        v16h n0 = *(const v16h*)(brown +  0);
        v16h n1 = *(const v16h*)(brown + 32);
        v16h n2 = *(const v16h*)(brown + 64);
        v16h n3 = *(const v16h*)(brown + 96);

        v8f c0 = {}, c1 = {};
        c0 = __builtin_amdgcn_wmma_f32_16x16x32_f16(false, a0, false, b0, (short)0, c0, false, false);
        c0 = __builtin_amdgcn_wmma_f32_16x16x32_f16(false, a1, false, b1, (short)0, c0, false, false);
        c0 = __builtin_amdgcn_wmma_f32_16x16x32_f16(false, a2, false, b2, (short)0, c0, false, false);
        c0 = __builtin_amdgcn_wmma_f32_16x16x32_f16(false, a3, false, b3, (short)0, c0, false, false);
        c1 = __builtin_amdgcn_wmma_f32_16x16x32_f16(false, a4, false, b0, (short)0, c1, false, false);
        c1 = __builtin_amdgcn_wmma_f32_16x16x32_f16(false, a5, false, b1, (short)0, c1, false, false);
        c1 = __builtin_amdgcn_wmma_f32_16x16x32_f16(false, a6, false, b2, (short)0, c1, false, false);
        c1 = __builtin_amdgcn_wmma_f32_16x16x32_f16(false, a7, false, b3, (short)0, c1, false, false);

        // Refresh register thresholds only after a compaction happened.
        if (needThr) {
            #pragma unroll
            for (int v = 0; v < 8; ++v) {
                thrv[v]     = thr[v + (g << 3)];
                thrv[v + 8] = thr[16 + v + (g << 3)];
            }
            needThr = false;
        }

        int  col   = colBase + lm;
        bool colOk = col < N;
        bool diag0 = (colBase == rowBase);
        bool diag1 = (colBase == rowBase + 16);

        #pragma unroll
        for (int v = 0; v < 8; ++v) {
            // low half: C layout M = v + 8*(lane/16)
            {
                int r = v + (g << 3);
                float val = c0[v] * TAU_INV;
                bool ok = colOk && rowOk[v];
                if (diag0 && ok && col == rowBase + r) { posv[r] = val; ok = false; }
                if (ok && val > thrv[v]) {
                    int q = atomicAdd(&cnt[r], 1);    // cap invariant: q < CAP
                    buf[r][q] = val;
                }
            }
            // high half
            {
                int r = 16 + v + (g << 3);
                float val = c1[v] * TAU_INV;
                bool ok = colOk && rowOk[v + 8];
                if (diag1 && ok && col == rowBase + r) { posv[r] = val; ok = false; }
                if (ok && val > thrv[v + 8]) {
                    int q = atomicAdd(&cnt[r], 1);
                    buf[r][q] = val;
                }
            }
        }
        __syncthreads();

        // Ballot-based compaction trigger: one LDS read + one vote per tile.
        unsigned long long m = __ballot(cnt[lane] > HARD_K);
        if (m) {
            needThr = true;                           // wave-uniform
            while (m) {
                int r = __ffsll(m) - 1;
                m &= m - 1;
                topk_compact(buf[r], &cnt[r], &thr[r], lane);
            }
        }
        __syncthreads();

        b0 = n0; b1 = n1; b2 = n2; b3 = n3;           // rotate double buffer
    }

    // Per-row logsumexp - pos; one row per lane (32 rows, 32 lanes).
    {
        int rowG = rowBase + lane;
        if (rowG < N) {
            int mcnt = cnt[lane];                     // == min(HARD_K, N-1)
            float pv = posv[lane];
            float mx = pv;
            for (int i = 0; i < mcnt; ++i) mx = fmaxf(mx, buf[lane][i]);
            float s = __expf(pv - mx);
            for (int i = 0; i < mcnt; ++i) s += __expf(buf[lane][i] - mx);
            float lse = mx + __logf(s);
            atomicAdd(&pairSum[p], lse - pv);
        }
    }
}

__global__ void k_final(const int* pairN, const float* pairSum, float* out) {
    if (threadIdx.x == 0 && blockIdx.x == 0) {
        const float w[3] = {0.2f, 1.0f, 1.0f};
        float total = 0.0f;
        for (int p = 0; p < 3; ++p) {
            int n = pairN[p];
            if (n >= 4) total += w[p] * (pairSum[p] / (float)n);  // MIN_OVERLAP gate + mean
        }
        out[0] = total;
    }
}

// ---------------- host launcher ----------------
extern "C" void kernel_launch(void* const* d_in, const int* in_sizes, int n_in,
                              void* d_out, int out_size, void* d_ws, size_t ws_size,
                              hipStream_t stream) {
    const float* embs[3] = { (const float*)d_in[0],        // view
                             (const float*)d_in[1],        // cart
                             (const float*)d_in[2] };      // purchase
    const long long* usrs[3] = { (const long long*)d_in[3],
                                 (const long long*)d_in[4],
                                 (const long long*)d_in[5] };
    int usrN[3] = { in_sizes[3], in_sizes[4], in_sizes[5] };

    char* ws = (char*)d_ws;
    unsigned char* bitmap   = (unsigned char*)(ws + OFF_BITMAP);
    unsigned int*  bm_words = (unsigned int*) (ws + OFF_BITMAP);
    int*           pairN    = (int*)          (ws + OFF_META_N);
    float*         pairSum  = (float*)        (ws + OFF_META_S);
    int*           common   = (int*)          (ws + OFF_COMMON);
    _Float16*      Aw       = (_Float16*)     (ws + OFF_AW);
    _Float16*      As       = (_Float16*)     (ws + OFF_AS);

    // pairs: (weak, strong): (view,cart), (cart,purchase), (view,purchase)
    const int pw[3] = {0, 1, 0};
    const int ps[3] = {1, 2, 2};

    const int bmWords = 100352 / 4;
    for (int p = 0; p < 3; ++p) {
        int nW = usrN[pw[p]], nS = usrN[ps[p]];
        k_zero<<<(bmWords + 255) / 256, 256, 0, stream>>>(bm_words, bmWords, pairN, pairSum, p);
        k_mark<<<(nS + 255) / 256, 256, 0, stream>>>(usrs[ps[p]], nS, bitmap);
        k_compact<<<(nW + 255) / 256, 256, 0, stream>>>(usrs[pw[p]], nW, bitmap, common, pairN, p);
        int gatherBlocks = ((nW + 31) / 32) * 32;
        k_gather<<<gatherBlocks, 128, 0, stream>>>(embs[pw[p]], embs[ps[p]], common, pairN, p, Aw, As);
        k_simloss<<<(nW + 31) / 32, 32, 0, stream>>>(Aw, As, pairN, pairSum, p);
    }
    k_final<<<1, 32, 0, stream>>>(pairN, pairSum, (float*)d_out);
}
